// GoldFinchTransformer_25391846654566
// MI455X (gfx1250) — matrix operand order repaired and checked by reading, not compiled, not run on hardware
//
// GoldFinch (RWKV-6-like) layer for MI455X / gfx1250, wave32, WMMA bf16 path.
// GEMMs use GLOBAL_LOAD_ASYNC_TO_LDS_B128 double-buffered tiles (ASYNCcnt
// pipelined) feeding v_wmma_f32_16x16x32_bf16. WKV recurrence is a 64-WG
// register-resident scan. Requires ws_size >= ~700 MB scratch.
#include <hip/hip_runtime.h>
#include <math.h>

typedef __bf16 bf16;
typedef __attribute__((ext_vector_type(16))) __bf16 v16bf;
typedef __attribute__((ext_vector_type(8)))  __bf16 v8bf;
typedef __attribute__((ext_vector_type(8)))  float  v8f;

static constexpr int Bsz = 4, T = 4096, C = 1024, NH = 16, DH = 64;
static constexpr int Mrows = Bsz * T;              // 16384
static constexpr size_t BTC = (size_t)Bsz * T * C; // 16777216

enum { EP_F32 = 0, EP_TANH_BF16 = 1, EP_ADD_F32 = 2, EP_BIAS_F32 = 3 };

__device__ __forceinline__ v16bf ld_frag(const bf16* p0, const bf16* p1) {
  v8bf lo = *(const v8bf*)p0;
  v8bf hi = *(const v8bf*)p1;
  return __builtin_shufflevector(lo, hi, 0,1,2,3,4,5,6,7,8,9,10,11,12,13,14,15);
}

// Async 16B global->LDS copy (no VGPR round trip); ASYNCcnt tracked.
// Low 32 bits of a generic LDS pointer are the LDS byte address (ISA 10.2).
__device__ __forceinline__ void async_ld16(unsigned ldsaddr, const bf16* gaddr) {
  asm volatile("global_load_async_to_lds_b128 %0, %1, off"
               :: "v"(ldsaddr), "v"(gaddr) : "memory");
}
template<int N>
__device__ __forceinline__ void wait_async() {
  asm volatile("s_wait_asynccnt %0" :: "i"(N) : "memory");
}

// ---------------------------------------------------------------- GEMM core
// C[M,N] = A[M,K](bf16, lda) x Bt[N,K](bf16, ldb), f32 accumulation.
// Block tile BM x BN, 8 waves, per-wave TM x TN WMMA 16x16 tiles, K step 32.
// Double-buffered async tile staging: tile t+1 in flight during compute(t),
// tile t+2 issued right after compute(t).
template<int BM, int BN, int WMv, int WNv, int TM, int TN, int EP>
__global__ __launch_bounds__(256) void k_gemm(
    const bf16* __restrict__ A, int lda,
    const bf16* __restrict__ Bt, int ldb,
    void* __restrict__ Cout, int ldc,
    int K, const float* __restrict__ bias)
{
  constexpr int LDSP = 40;           // 32 elems + pad (80B rows, 16B aligned)
  constexpr int ACH = BM * 4;        // 16B chunks per A tile
  constexpr int BCH = BN * 4;
  constexpr int AIT = ACH / 256;     // chunks per thread (BM=128 -> 2)
  constexpr int BIT = (BCH + 255) / 256;
  constexpr int IT  = AIT + BIT;     // async issues per wave per tile (uniform)
  static_assert(ACH % 256 == 0, "BM tile must fill threads evenly");

  __shared__ bf16 sA[2][BM * LDSP];
  __shared__ bf16 sB[2][BN * LDSP];
  constexpr unsigned ASTRIDE = BM * LDSP * 2;  // bytes between A buffers
  constexpr unsigned BSTRIDE = BN * LDSP * 2;

  const int tid  = threadIdx.x;
  const int wave = tid >> 5;
  const int lane = tid & 31;
  const int l    = lane & 15;
  const int hi   = lane >> 4;        // lane-half select per WMMA layout
  const int wm   = wave % WMv;
  const int wn   = wave / WMv;
  const int mBase = blockIdx.y * BM;
  const int nBase = blockIdx.x * BN;

  // Hoisted per-thread chunk descriptors (no per-K-step 64-bit muls).
  const bf16* aSrc[AIT]; unsigned aLds[AIT];
  #pragma unroll
  for (int ci = 0; ci < AIT; ++ci) {
    const int idx = tid + ci * 256, r = idx >> 2, o = (idx & 3) * 8;
    aSrc[ci] = A + (size_t)(mBase + r) * lda + o;
    aLds[ci] = (unsigned)(uintptr_t)&sA[0][r * LDSP + o];
  }
  const bf16* bSrc[BIT]; unsigned bLds[BIT];
  #pragma unroll
  for (int ci = 0; ci < BIT; ++ci) {
    // modulo keeps every lane active (dup writes of identical data are benign)
    const int idx = (tid + ci * 256) % BCH, r = idx >> 2, o = (idx & 3) * 8;
    bSrc[ci] = Bt + (size_t)(nBase + r) * ldb + o;
    bLds[ci] = (unsigned)(uintptr_t)&sB[0][r * LDSP + o];
  }

  const int NT = K / 32;
  auto issue = [&](int t, int buf) {
    const int kb = t * 32;
    #pragma unroll
    for (int ci = 0; ci < AIT; ++ci)
      async_ld16(aLds[ci] + buf * ASTRIDE, aSrc[ci] + kb);
    #pragma unroll
    for (int ci = 0; ci < BIT; ++ci)
      async_ld16(bLds[ci] + buf * BSTRIDE, bSrc[ci] + kb);
  };

  v8f acc[TM][TN] = {};

  issue(0, 0);
  if (NT > 1) issue(1, 1);

  for (int it = 0; it < NT; ++it) {
    const int buf = it & 1;
    if (it + 1 < NT) wait_async<IT>();  // tile `it` landed; `it+1` in flight
    else             wait_async<0>();
    __syncthreads();                    // all waves' tile `it` visible

    v16bf af[TM], bfr[TN];
    #pragma unroll
    for (int i = 0; i < TM; ++i) {
      // A 16x32 bf16: lane<16 row l holds K{0..7,16..23}; lane>=16 K{8..15,24..31}
      const bf16* p = &sA[buf][((wm * TM + i) * 16 + l) * LDSP + hi * 8];
      af[i] = ld_frag(p, p + 16);
    }
    #pragma unroll
    for (int j = 0; j < TN; ++j) {
      // B 32x16 bf16: lane<16 col l holds K{0..7,16..23}; lane>=16 K{8..15,24..31}
      const bf16* p = &sB[buf][((wn * TN + j) * 16 + l) * LDSP + hi * 8];
      bfr[j] = ld_frag(p, p + 16);
    }
    #pragma unroll
    for (int i = 0; i < TM; ++i)
      #pragma unroll
      for (int j = 0; j < TN; ++j)
        acc[i][j] = __builtin_amdgcn_wmma_f32_16x16x32_bf16(
            false, af[i], false, bfr[j], (short)0, acc[i][j], false, false);

    __syncthreads();                    // compute(it) done everywhere
    if (it + 2 < NT) issue(it + 2, buf);// safe to overwrite buf now
  }

  #pragma unroll
  for (int i = 0; i < TM; ++i) {
    const int row0 = mBase + (wm * TM + i) * 16 + hi * 8;
    #pragma unroll
    for (int j = 0; j < TN; ++j) {
      const int col = nBase + (wn * TN + j) * 16 + l;
      #pragma unroll
      for (int e = 0; e < 8; ++e) {
        const float v = acc[i][j][e];
        const size_t idx = (size_t)(row0 + e) * ldc + col;
        if constexpr (EP == EP_F32)            ((float*)Cout)[idx] = v;
        else if constexpr (EP == EP_TANH_BF16) ((bf16*)Cout)[idx] = (bf16)tanhf(v);
        else if constexpr (EP == EP_ADD_F32)   ((float*)Cout)[idx] += v;
        else                                   ((float*)Cout)[idx] = v + bias[col];
      }
    }
  }
}

// ----------------------------------------------- weight transpose+cast (f32->bf16)
// out[c*R + r] = in[r*Cc + c];  R,Cc multiples of 32 here.
__global__ __launch_bounds__(256) void k_transpose_cast(
    const float* __restrict__ in, bf16* __restrict__ out, int R, int Cc)
{
  __shared__ float tile[32][33];
  const int c0 = blockIdx.x * 32, r0 = blockIdx.y * 32;
  const int tx = threadIdx.x, ty = threadIdx.y; // 32 x 8
  #pragma unroll
  for (int i = 0; i < 4; ++i) {
    int r = r0 + ty + i * 8, c = c0 + tx;
    if (r < R && c < Cc) tile[ty + i * 8][tx] = in[(size_t)r * Cc + c];
  }
  __syncthreads();
  #pragma unroll
  for (int i = 0; i < 4; ++i) {
    int c = c0 + ty + i * 8, r = r0 + tx;
    if (r < R && c < Cc) out[(size_t)c * R + r] = (bf16)tile[tx][ty + i * 8];
  }
}

// ------------------------------------- token shift: dxprev, xxx = x + dx*maa_x
__global__ __launch_bounds__(256) void k_prep(
    const float* __restrict__ x, const float* __restrict__ shift,
    const float* __restrict__ tmx, float* __restrict__ dxprev,
    bf16* __restrict__ xxx)
{
  const size_t i = (size_t)blockIdx.x * 256 + threadIdx.x;
  const int c = (int)(i & (C - 1));
  const size_t bt = i >> 10;
  const int t = (int)(bt & (T - 1));
  const int b = (int)(bt >> 12);
  const float xv = x[i];
  const float prev = (t == 0) ? shift[(size_t)b * C + c] : x[i - C];
  const float d = prev - xv;
  dxprev[i] = d;
  xxx[i] = (bf16)(xv + d * tmx[c]);
}

// ---------------- xf = bf16( x + dxprev * (time_maa_f[c] + m[i]) ) per branch
__global__ __launch_bounds__(256) void k_combine(
    const float* __restrict__ x, const float* __restrict__ dxprev,
    const float* __restrict__ tm, const float* __restrict__ m,
    bf16* __restrict__ out)
{
  const size_t i = (size_t)blockIdx.x * 256 + threadIdx.x;
  const int c = (int)(i & (C - 1));
  out[i] = (bf16)(x[i] + dxprev[i] * (tm[c] + m[i]));
}

// ------------------------------------------------------------- WKV recurrence
// One workgroup per (b, head). Thread (g=tid/64, c=tid&63) owns S[k=g*16+i][c].
// decay = exp(-exp(w)); k <- k*(1-decay) folded into staging.
__global__ __launch_bounds__(256) void k_scan(
    const float* __restrict__ r, const float* __restrict__ k,
    const float* __restrict__ v, const float* __restrict__ v2,
    const float* __restrict__ w, const float* __restrict__ wkv0,
    float* __restrict__ y, float* __restrict__ fstate)
{
  __shared__ float sh_d[64], sh_k[64], sh_v[64], sh_r[64], sh_part[256];
  const int b = blockIdx.x >> 4, n = blockIdx.x & 15;
  const int tid = threadIdx.x, g = tid >> 6, c = tid & 63;
  const size_t sbase = (size_t)blockIdx.x * DH * DH;

  float S[16];
  #pragma unroll
  for (int i = 0; i < 16; ++i)
    S[i] = wkv0[sbase + (size_t)(g * 16 + i) * DH + c];

  for (int t = 0; t < T; ++t) {
    const size_t ro = ((size_t)b * T + t) * C + n * DH;
    if (tid < 64)       sh_d[tid]       = __expf(-__expf(w[ro + tid]));
    else if (tid < 128) sh_v[tid - 64]  = v[ro + tid - 64];
    else if (tid < 192) sh_r[tid - 128] = r[ro + tid - 128];
    else                sh_k[tid - 192] = k[ro + tid - 192];
    __syncthreads();
    if (tid < 64) sh_k[tid] *= (1.0f - sh_d[tid]);
    __syncthreads();

    float accv = 0.0f;
    const float vc = sh_v[c];
    #pragma unroll
    for (int i = 0; i < 16; ++i) {
      const int kk = g * 16 + i;
      const float s = S[i];
      accv = fmaf(sh_r[kk], s, accv);          // y uses pre-update state
      S[i] = fmaf(s, sh_d[kk], sh_k[kk] * vc); // state = state*decay + k~ (x) v
    }
    sh_part[tid] = accv;
    __syncthreads();
    if (tid < 64)
      y[ro + c] = sh_part[c] + sh_part[64 + c] + sh_part[128 + c] +
                  sh_part[192 + c] + v2[ro + c];
    __syncthreads();
  }
  #pragma unroll
  for (int i = 0; i < 16; ++i)
    fstate[sbase + (size_t)(g * 16 + i) * DH + c] = S[i];
}

// ----------------------------------------------------- LayerNorm -> bf16 rows
__global__ __launch_bounds__(256) void k_layernorm(
    const float* __restrict__ y, const float* __restrict__ gam,
    const float* __restrict__ bet, bf16* __restrict__ out)
{
  __shared__ float red[256];
  const int row = blockIdx.x, tid = threadIdx.x;
  const float4 xv = ((const float4*)(y + (size_t)row * C))[tid];
  red[tid] = xv.x + xv.y + xv.z + xv.w;
  __syncthreads();
  for (int o = 128; o > 0; o >>= 1) { if (tid < o) red[tid] += red[tid + o]; __syncthreads(); }
  const float mu = red[0] * (1.0f / C);
  __syncthreads();
  const float d0 = xv.x - mu, d1 = xv.y - mu, d2 = xv.z - mu, d3 = xv.w - mu;
  red[tid] = d0 * d0 + d1 * d1 + d2 * d2 + d3 * d3;
  __syncthreads();
  for (int o = 128; o > 0; o >>= 1) { if (tid < o) red[tid] += red[tid + o]; __syncthreads(); }
  const float rstd = rsqrtf(red[0] * (1.0f / C) + 1e-5f);
  const float4 gv = ((const float4*)gam)[tid];
  const float4 bv = ((const float4*)bet)[tid];
  bf16* o4 = out + (size_t)row * C + tid * 4;
  o4[0] = (bf16)(d0 * rstd * gv.x + bv.x);
  o4[1] = (bf16)(d1 * rstd * gv.y + bv.y);
  o4[2] = (bf16)(d2 * rstd * gv.z + bv.z);
  o4[3] = (bf16)(d3 * rstd * gv.w + bv.w);
}

// =============================================================== host driver
extern "C" void kernel_launch(void* const* d_in, const int* in_sizes, int n_in,
                              void* d_out, int out_size, void* d_ws, size_t ws_size,
                              hipStream_t stream) {
  (void)in_sizes; (void)n_in; (void)out_size; (void)ws_size;
  const float* x       = (const float*)d_in[0];
  const float* shiftst = (const float*)d_in[1];
  const float* wkv0    = (const float*)d_in[2];
  const float* tmx     = (const float*)d_in[3];
  const float* tmr     = (const float*)d_in[4];
  const float* tmk     = (const float*)d_in[5];
  const float* tmv     = (const float*)d_in[6];
  const float* tmw     = (const float*)d_in[7];
  const float* tmv2    = (const float*)d_in[8];
  const float* maa_w1  = (const float*)d_in[9];   // (C,160)
  const float* maa_w2  = (const float*)d_in[10];  // (5,32,C)
  const float* tdecay  = (const float*)d_in[11];  // (C)
  const float* tdw1    = (const float*)d_in[12];  // (C,64)
  const float* tdw2    = (const float*)d_in[13];  // (64,C)
  const float* tv2w1   = (const float*)d_in[14];  // (C,64)
  const float* tv2w2   = (const float*)d_in[15];  // (64,C)
  const float* W_r     = (const float*)d_in[16];
  const float* W_k     = (const float*)d_in[17];
  const float* W_v     = (const float*)d_in[18];
  const float* W_o     = (const float*)d_in[19];
  const float* ln_g    = (const float*)d_in[20];
  const float* ln_b    = (const float*)d_in[21];

  float* y_out  = (float*)d_out;          // (B,T,C)
  float* fstate = (float*)d_out + BTC;    // (B,N,64,64)

  // workspace carve-up (bump allocator, 256B aligned)
  char* wp = (char*)d_ws;
  auto alloc = [&](size_t bytes) -> void* {
    void* r = (void*)wp;
    wp += (bytes + 255) & ~(size_t)255;
    return r;
  };
  bf16* Wrt    = (bf16*)alloc((size_t)C * C * 2);
  bf16* Wkt    = (bf16*)alloc((size_t)C * C * 2);
  bf16* Wvt    = (bf16*)alloc((size_t)C * C * 2);
  bf16* Wot    = (bf16*)alloc((size_t)C * C * 2);
  bf16* W1t    = (bf16*)alloc((size_t)160 * C * 2);
  bf16* W2t    = (bf16*)alloc((size_t)5 * C * 32 * 2);
  bf16* tdw1t  = (bf16*)alloc((size_t)64 * C * 2);
  bf16* tdw2t  = (bf16*)alloc((size_t)C * 64 * 2);
  bf16* tv2w1t = (bf16*)alloc((size_t)64 * C * 2);
  bf16* tv2w2t = (bf16*)alloc((size_t)C * 64 * 2);
  float* dxprev = (float*)alloc(BTC * 4);
  bf16*  xxx    = (bf16*)alloc(BTC * 2);          // reused later as y_ln
  bf16*  mixed  = (bf16*)alloc((size_t)Mrows * 160 * 2);
  float* mbuf   = (float*)alloc(BTC * 4);         // reused later as y
  bf16*  xr     = (bf16*)alloc(BTC * 2);
  bf16*  xk     = (bf16*)alloc(BTC * 2);
  bf16*  xv     = (bf16*)alloc(BTC * 2);
  bf16*  xw     = (bf16*)alloc(BTC * 2);
  bf16*  xv2    = (bf16*)alloc(BTC * 2);
  float* rb     = (float*)alloc(BTC * 4);
  float* kb     = (float*)alloc(BTC * 4);
  float* vb     = (float*)alloc(BTC * 4);
  float* v2b    = (float*)alloc(BTC * 4);
  float* wb     = (float*)alloc(BTC * 4);
  bf16*  t2     = (bf16*)alloc((size_t)Mrows * 64 * 2);
  bf16*  t3     = (bf16*)alloc((size_t)Mrows * 64 * 2);

  const dim3 tb(32, 8);
  auto tgrid = [](int R, int Cc) { return dim3((Cc + 31) / 32, (R + 31) / 32); };

  // 1) transpose-cast all GEMM B operands to bf16 [N][K]
  k_transpose_cast<<<tgrid(C, C), tb, 0, stream>>>(W_r, Wrt, C, C);
  k_transpose_cast<<<tgrid(C, C), tb, 0, stream>>>(W_k, Wkt, C, C);
  k_transpose_cast<<<tgrid(C, C), tb, 0, stream>>>(W_v, Wvt, C, C);
  k_transpose_cast<<<tgrid(C, C), tb, 0, stream>>>(W_o, Wot, C, C);
  k_transpose_cast<<<tgrid(C, 160), tb, 0, stream>>>(maa_w1, W1t, C, 160);
  for (int f = 0; f < 5; ++f)
    k_transpose_cast<<<tgrid(32, C), tb, 0, stream>>>(
        maa_w2 + (size_t)f * 32 * C, W2t + (size_t)f * C * 32, 32, C);
  k_transpose_cast<<<tgrid(C, 64), tb, 0, stream>>>(tdw1, tdw1t, C, 64);
  k_transpose_cast<<<tgrid(64, C), tb, 0, stream>>>(tdw2, tdw2t, 64, C);
  k_transpose_cast<<<tgrid(C, 64), tb, 0, stream>>>(tv2w1, tv2w1t, C, 64);
  k_transpose_cast<<<tgrid(64, C), tb, 0, stream>>>(tv2w2, tv2w2t, 64, C);

  const int EW = (int)(BTC / 256);       // elementwise grid (65536 blocks)

  // 2) token shift
  k_prep<<<EW, 256, 0, stream>>>(x, shiftst, tmx, dxprev, xxx);

  // 3) mixed = tanh(xxx @ W1)  (M=16384, K=1024, N=160) -> bf16
  k_gemm<128, 32, 8, 1, 1, 2, EP_TANH_BF16><<<dim3(160 / 32, Mrows / 128), 256, 0, stream>>>(
      xxx, C, W1t, C, mixed, 160, C, nullptr);

  // 4) per-branch m_f = mixed[:, f] @ W2_f, then xf = x + dx*(maa_f + m_f)
  const float* tmaas[5] = {tmr, tmk, tmv, tmw, tmv2};
  bf16* xfs[5] = {xr, xk, xv, xw, xv2};
  for (int f = 0; f < 5; ++f) {
    k_gemm<128, 128, 4, 2, 2, 4, EP_F32><<<dim3(C / 128, Mrows / 128), 256, 0, stream>>>(
        mixed + f * 32, 160, W2t + (size_t)f * C * 32, 32, mbuf, C, 32, nullptr);
    k_combine<<<EW, 256, 0, stream>>>(x, dxprev, tmaas[f], mbuf, xfs[f]);
  }

  // 5) projections (the 172-GFLOP WMMA bulk)
  k_gemm<128, 128, 4, 2, 2, 4, EP_F32><<<dim3(C / 128, Mrows / 128), 256, 0, stream>>>(
      xr, C, Wrt, C, rb, C, C, nullptr);
  k_gemm<128, 128, 4, 2, 2, 4, EP_F32><<<dim3(C / 128, Mrows / 128), 256, 0, stream>>>(
      xk, C, Wkt, C, kb, C, C, nullptr);
  k_gemm<128, 128, 4, 2, 2, 4, EP_F32><<<dim3(C / 128, Mrows / 128), 256, 0, stream>>>(
      xv, C, Wvt, C, vb, C, C, nullptr);
  k_gemm<128, 128, 4, 2, 2, 4, EP_F32><<<dim3(C / 128, Mrows / 128), 256, 0, stream>>>(
      xv2, C, Wvt, C, v2b, C, C, nullptr);
  // v2 += tanh(xv2 @ tv2w1) @ tv2w2
  k_gemm<128, 32, 8, 1, 1, 2, EP_TANH_BF16><<<dim3(64 / 32, Mrows / 128), 256, 0, stream>>>(
      xv2, C, tv2w1t, C, t2, 64, C, nullptr);
  k_gemm<128, 128, 4, 2, 2, 4, EP_ADD_F32><<<dim3(C / 128, Mrows / 128), 256, 0, stream>>>(
      t2, 64, tv2w2t, 64, v2b, C, 64, nullptr);
  // w = time_decay + tanh(xw @ tdw1) @ tdw2
  k_gemm<128, 32, 8, 1, 1, 2, EP_TANH_BF16><<<dim3(64 / 32, Mrows / 128), 256, 0, stream>>>(
      xw, C, tdw1t, C, t3, 64, C, nullptr);
  k_gemm<128, 128, 4, 2, 2, 4, EP_BIAS_F32><<<dim3(C / 128, Mrows / 128), 256, 0, stream>>>(
      t3, 64, tdw2t, 64, wb, C, 64, tdecay);

  // 6) sequential WKV scan: y (pre-LN) into mbuf, final state straight to d_out
  float* ybuf = mbuf;
  k_scan<<<Bsz * NH, 256, 0, stream>>>(rb, kb, vb, v2b, wb, wkv0, ybuf, fstate);

  // 7) LayerNorm -> bf16 (reuse xxx), then output projection into d_out
  bf16* y_ln = xxx;
  k_layernorm<<<Mrows, 256, 0, stream>>>(ybuf, ln_g, ln_b, y_ln);
  k_gemm<128, 128, 4, 2, 2, 4, EP_F32><<<dim3(C / 128, Mrows / 128), 256, 0, stream>>>(
      y_ln, C, Wot, C, y_out, C, C, nullptr);
}
